// MaskEnc_37495064494385
// MI455X (gfx1250) — matrix operand order repaired and checked by reading, not compiled, not run on hardware
//
#include <hip/hip_runtime.h>
#include <hip/hip_bf16.h>

// ---------------------------------------------------------------------------
// MaskEnc transformer layer for MI455X (gfx1250, wave32, WMMA).
// All matmuls (QKV, QK^T, PV, FFN1, FFN2) use v_wmma_f32_16x16x32_f16.
// A-tile staging uses gfx1250 async global->LDS DMA (ASYNCcnt).
// ---------------------------------------------------------------------------

typedef __attribute__((ext_vector_type(16))) _Float16 v16h;
typedef __attribute__((ext_vector_type(8)))  _Float16 v8h;
typedef __attribute__((ext_vector_type(8)))  float    v8f;

#define B_   4
#define T_   1024
#define D_   1024
#define H_   16
#define DH_  64
#define NROW 4096            // B*T
#define NEG_BIG (-4294967295.0f)

// Flip to 0 if the assembler rejects the async mnemonic.
#define USE_ASYNC_LDS 1

static __device__ __forceinline__ v16h cat8(v8h lo, v8h hi) {
  return __builtin_shufflevector(lo, hi, 0,1,2,3,4,5,6,7,8,9,10,11,12,13,14,15);
}
static __device__ __forceinline__ v8f wmma16(v16h a, v16h b, v8f c) {
  return __builtin_amdgcn_wmma_f32_16x16x32_f16(false, a, false, b, (short)0, c,
                                                false, false);
}

#if USE_ASYNC_LDS
// Async DMA: 16B per lane, global -> LDS, tracked by ASYNCcnt.
// Generic pointers to LDS have the LDS byte offset in their low 32 bits
// (flat-LDS truncation rule, CDNA5 ISA).
static __device__ __forceinline__ void async_g2l_b128(void* lds,
                                                      const void* gptr) {
  unsigned laddr = (unsigned)(unsigned long long)lds;
  asm volatile("global_load_async_to_lds_b128 %0, %1, off"
               :: "v"(laddr), "v"(gptr) : "memory");
}
static __device__ __forceinline__ void wait_async0() {
  asm volatile("s_wait_asynccnt 0x0" ::: "memory");
}
#endif

// ------------------------------- f32 -> f16 --------------------------------
__global__ __launch_bounds__(256) void cvt_f32_f16(const float* __restrict__ in,
                                                   _Float16* __restrict__ out) {
  int i = blockIdx.x * 256 + threadIdx.x;
  out[i] = (_Float16)in[i];
}

// --------------------------------- GEMM ------------------------------------
// C[4096x1024] = act(A_f16[4096x1024] @ W_f16[1024x1024] + bias)
// Block tile 64x128, BK=64, 256 thr = 8 waves (2 M-halves x 4 N-quarters),
// each wave 32x32 -> 2x2 tiles x 2 K-steps = 8 WMMA per barrier pair.
// W staged transposed in LDS so B fragments are contiguous ds_load_b128
// pairs (16-bit A/B layouts, ISA 7.12.2).  Epilogue fully templated.
template <bool HAS_BIAS, bool RELU, bool OUT32, bool OUT16>
__global__ __launch_bounds__(256) void gemm_kernel(
    const _Float16* __restrict__ A, const _Float16* __restrict__ W,
    const float* __restrict__ bias, float* __restrict__ Cf,
    _Float16* __restrict__ Ch) {
  __shared__ _Float16 As[64][72];    // 64 rows x BK(64)+8 pad  (9.2 KB)
  __shared__ _Float16 Bt[128][72];   // 128 cols x BK(64)+8 pad (18.4 KB)

  const int tid  = threadIdx.x;
  const int lane = tid & 31, wid = tid >> 5;
  const int wm = wid & 1, wn = wid >> 1;     // wave position in 64x128 tile
  const int g  = lane >> 4, ln = lane & 15;  // half-wave group / sub-lane
  const int m0 = blockIdx.y * 64, n0 = blockIdx.x * 128;

  v8f zero = {};
  v8f acc[2][2] = {{zero, zero}, {zero, zero}};

  for (int k0 = 0; k0 < 1024; k0 += 64) {
    // ---- stage A (64x64) via async DMA, W^T (64x128) via LDS transpose ----
#pragma unroll
    for (int i = 0; i < 2; i++) {            // A: 512 x 16B chunks
      int idx = tid + i * 256;
      int row = idx >> 3, seg = idx & 7;
      const _Float16* ap = A + (size_t)(m0 + row) * 1024 + k0 + seg * 8;
#if USE_ASYNC_LDS
      async_g2l_b128(&As[row][seg * 8], ap);
#else
      *(v8h*)&As[row][seg * 8] = *(const v8h*)ap;
#endif
    }
#pragma unroll
    for (int i = 0; i < 4; i++) {            // W: 1024 x 16B chunks, transpose
      int idx = tid + i * 256;
      int kk = idx >> 4, sg = idx & 15;
      const _Float16* wp = W + (size_t)(k0 + kk) * 1024 + n0 + sg * 8;
      v8h bv = *(const v8h*)wp;
      if (k0 + 64 < 1024) __builtin_prefetch(wp + 64 * 1024, 0, 1);
#pragma unroll
      for (int j = 0; j < 8; j++) Bt[sg * 8 + j][kk] = bv[j];
    }
#if USE_ASYNC_LDS
    wait_async0();
#endif
    __syncthreads();

    // ---- 2 K-steps x 2x2 tiles: 8 WMMA per stage ----
#pragma unroll
    for (int ks = 0; ks < 2; ks++) {
      v16h af[2], bf[2];
#pragma unroll
      for (int mt = 0; mt < 2; mt++) {
        int r = wm * 32 + mt * 16 + ln;
        af[mt] = cat8(*(const v8h*)&As[r][ks * 32 + 16 * g],
                      *(const v8h*)&As[r][ks * 32 + 16 * g + 8]);
      }
#pragma unroll
      for (int nt = 0; nt < 2; nt++) {
        int c = wn * 32 + nt * 16 + ln;
        bf[nt] = cat8(*(const v8h*)&Bt[c][ks * 32 + 16 * g],
                      *(const v8h*)&Bt[c][ks * 32 + 16 * g + 8]);
      }
#pragma unroll
      for (int mt = 0; mt < 2; mt++)
#pragma unroll
        for (int nt = 0; nt < 2; nt++)
          acc[mt][nt] = wmma16(af[mt], bf[nt], acc[mt][nt]);
    }
    __syncthreads();
  }

  // ---- branch-free epilogue: lane=col, vgpr v -> row v + 8*(lane/16) ----
#pragma unroll
  for (int mt = 0; mt < 2; mt++)
#pragma unroll
    for (int nt = 0; nt < 2; nt++) {
      int c = n0 + wn * 32 + nt * 16 + ln;
      float bb = HAS_BIAS ? bias[c] : 0.f;
#pragma unroll
      for (int vg = 0; vg < 8; vg++) {
        int r = m0 + wm * 32 + mt * 16 + vg + 8 * g;
        float val = acc[mt][nt][vg] + bb;
        if (RELU) val = fmaxf(val, 0.f);
        size_t o = (size_t)r * 1024 + c;
        if (OUT32) Cf[o] = val;
        if (OUT16) Ch[o] = (_Float16)val;
      }
    }
}

// ------------------------------ Attention ----------------------------------
// One block = 32 query rows of one (b,h).  Score strip lives in LDS
// (32 x 1024 f32, stride 1026 for conflict-free row access), softmax done
// in-wave, P converted f16 in place (low 16b of each f32 slot), then P@V.
#define SLP 1026
__global__ __launch_bounds__(256) void attn_kernel(
    const _Float16* __restrict__ Qh, const _Float16* __restrict__ Kh,
    const _Float16* __restrict__ Vh, float* __restrict__ O) {
  __shared__ float    Sl[32 * SLP];      // 131 KB score strip
  __shared__ _Float16 Qs[32][72];        // Q tile, padded
  __shared__ _Float16 Vt[64][40];        // V^T staging tile, padded
  __shared__ float    km[1024];          // key mask flags
  __shared__ float    qm[32];            // query mask flags

  const int tid = threadIdx.x;
  const int lane = tid & 31, wid = tid >> 5;
  const int g = lane >> 4, ln = lane & 15;
  const int bh = blockIdx.y, b = bh >> 4, h = bh & 15;
  const int q0 = blockIdx.x * 32;
  const size_t hoff = (size_t)h * DH_;

  // ---- phase 0: load Q tile, key/query masks ----
  {
    int row = tid >> 3, seg = tid & 7;
    *(v8h*)&Qs[row][seg * 8] =
        *(const v8h*)(Qh + ((size_t)(b * T_ + q0 + row) * D_) + hoff + seg * 8);
  }
  for (int key = tid; key < T_; key += 256) {
    const _Float16* kr = Kh + ((size_t)(b * T_ + key) * D_) + hoff;
    float s = 0.f;
    for (int j = 0; j < DH_; j++) s += fabsf((float)kr[j]);
    km[key] = (s != 0.f) ? 1.f : 0.f;
  }
  __syncthreads();
  if (tid < 32) {
    float s = 0.f;
    for (int j = 0; j < DH_; j++) s += fabsf((float)Qs[tid][j]);
    qm[tid] = (s != 0.f) ? 1.f : 0.f;
  }
  __syncthreads();

  // ---- phase 1: S = Q K^T * 1/8, + key mask + strict-future mask ----
  for (int nt = 0; nt < 8; nt++) {
    int kb = wid * 128 + nt * 16;
    int key = kb + ln;
    v16h bfr[2];  // B fragment: lane=key col, k contiguous in K's row-major
#pragma unroll
    for (int ks = 0; ks < 2; ks++)
      bfr[ks] = *(const v16h*)(Kh + ((size_t)(b * T_ + key) * D_) + hoff +
                               ks * 32 + 16 * g);
#pragma unroll
    for (int mt = 0; mt < 2; mt++) {
      v8f acc = {};
#pragma unroll
      for (int ks = 0; ks < 2; ks++) {
        int r = mt * 16 + ln;
        v16h a = cat8(*(const v8h*)&Qs[r][ks * 32 + 16 * g],
                      *(const v8h*)&Qs[r][ks * 32 + 16 * g + 8]);
        acc = wmma16(a, bfr[ks], acc);
      }
#pragma unroll
      for (int vg = 0; vg < 8; vg++) {
        int rl = mt * 16 + vg + 8 * g;
        int qrow = q0 + rl;
        float s = acc[vg] * 0.125f;           // 1/sqrt(64)
        if (km[key] == 0.f) s = NEG_BIG;      // key padding mask
        if (!(key > qrow)) s += -10000.0f;    // FORWARD strict mask
        Sl[rl * SLP + key] = s;
      }
    }
  }
  __syncthreads();

  // ---- phase 2: softmax per row + in-place f16 conversion (P) ----
  _Float16* Sh = (_Float16*)Sl;
  for (int i = 0; i < 4; i++) {
    int r = wid + 8 * i;
    float lm = -3.4e38f;
    for (int c = lane; c < T_; c += 32) lm = fmaxf(lm, Sl[r * SLP + c]);
    for (int o = 16; o; o >>= 1) lm = fmaxf(lm, __shfl_xor(lm, o, 32));
    float ls = 0.f;
    for (int c = lane; c < T_; c += 32) ls += __expf(Sl[r * SLP + c] - lm);
    for (int o = 16; o; o >>= 1) ls += __shfl_xor(ls, o, 32);
    float inv = qm[r] / ls;                   // fold post-softmax query mask
    for (int c = lane; c < T_; c += 32) {
      float p = __expf(Sl[r * SLP + c] - lm) * inv;
      Sh[(r * SLP + c) * 2] = (_Float16)p;    // low 16b of same slot: no alias
    }
  }
  __syncthreads();

  // ---- phase 3: O = P @ V  (8 waves -> 2x4 grid of 16x16 tiles) ----
  const int mt = wid & 1, nt = wid >> 1;
  v8f acc = {};
  for (int kc = 0; kc < T_; kc += 32) {
    __syncthreads();
    {   // stage V^T tile (32 keys x 64 dh)
      int kk = tid >> 3, seg = tid & 7;
      v8h vv = *(const v8h*)(Vh + ((size_t)(b * T_ + kc + kk) * D_) + hoff +
                             seg * 8);
#pragma unroll
      for (int j = 0; j < 8; j++) Vt[seg * 8 + j][kk] = vv[j];
    }
    __syncthreads();
    v16h a;
    int r = mt * 16 + ln;
#pragma unroll
    for (int e = 0; e < 16; e++) {  // A element e -> k = e%8 + 16*(e/8) + 8*g
      int k = kc + (e & 7) + 16 * (e >> 3) + 8 * g;
      a[e] = Sh[(r * SLP + k) * 2];
    }
    int c = nt * 16 + ln;
    v16h bv = cat8(*(const v8h*)&Vt[c][8 * g], *(const v8h*)&Vt[c][16 + 8 * g]);
    acc = wmma16(a, bv, acc);
  }
#pragma unroll
  for (int vg = 0; vg < 8; vg++) {
    int rl = mt * 16 + vg + 8 * g;
    int col = nt * 16 + ln;
    O[((size_t)(b * T_ + q0 + rl) * D_) + hoff + col] = acc[vg];
  }
}

// --------------------------- LayerNorm kernels -----------------------------
static __device__ __forceinline__ float block_sum(float v, float* red) {
  int lane = threadIdx.x & 31, wid = threadIdx.x >> 5;
  for (int o = 16; o; o >>= 1) v += __shfl_xor(v, o, 32);
  if (lane == 0) red[wid] = v;
  __syncthreads();
  float t = 0.f;
#pragma unroll
  for (int i = 0; i < 8; i++) t += red[i];
  __syncthreads();
  return t;
}

// h1 = LN(x + attn_out * token_mask); emit f32 + f16
__global__ __launch_bounds__(256) void ln1_kernel(
    const float* __restrict__ x, const float* __restrict__ attn,
    const float* __restrict__ mask, const float* __restrict__ gam,
    const float* __restrict__ bet, float* __restrict__ h1,
    _Float16* __restrict__ h1h) {
  __shared__ float red[8];
  int row = blockIdx.x, tid = threadIdx.x;
  float m = mask[row];
  float v[4], s = 0.f;
#pragma unroll
  for (int i = 0; i < 4; i++) {
    size_t o = (size_t)row * 1024 + tid + 256 * i;
    v[i] = x[o] + attn[o] * m;
    s += v[i];
  }
  float mean = block_sum(s, red) * (1.f / 1024.f);
  float var = 0.f;
#pragma unroll
  for (int i = 0; i < 4; i++) { float d = v[i] - mean; var += d * d; }
  var = block_sum(var, red) * (1.f / 1024.f);
  float rstd = rsqrtf(var + 1e-5f);
#pragma unroll
  for (int i = 0; i < 4; i++) {
    int idx = tid + 256 * i;
    size_t o = (size_t)row * 1024 + idx;
    float y = (v[i] - mean) * rstd * gam[idx] + bet[idx];
    h1[o] = y;
    h1h[o] = (_Float16)y;
  }
}

// out = LN3(LN2(h1 + ffn2))  (fused double layernorm)
__global__ __launch_bounds__(256) void ln23_kernel(
    const float* __restrict__ h1, const float* __restrict__ f2,
    const float* __restrict__ g2, const float* __restrict__ b2,
    const float* __restrict__ g3, const float* __restrict__ b3,
    float* __restrict__ out) {
  __shared__ float red[8];
  int row = blockIdx.x, tid = threadIdx.x;
  float v[4], s = 0.f;
#pragma unroll
  for (int i = 0; i < 4; i++) {
    size_t o = (size_t)row * 1024 + tid + 256 * i;
    v[i] = h1[o] + f2[o];
    s += v[i];
  }
  float mean = block_sum(s, red) * (1.f / 1024.f);
  float var = 0.f;
#pragma unroll
  for (int i = 0; i < 4; i++) { float d = v[i] - mean; var += d * d; }
  var = block_sum(var, red) * (1.f / 1024.f);
  float rstd = rsqrtf(var + 1e-5f);
  float y[4]; s = 0.f;
#pragma unroll
  for (int i = 0; i < 4; i++) {
    int idx = tid + 256 * i;
    y[i] = (v[i] - mean) * rstd * g2[idx] + b2[idx];
    s += y[i];
  }
  mean = block_sum(s, red) * (1.f / 1024.f);
  var = 0.f;
#pragma unroll
  for (int i = 0; i < 4; i++) { float d = y[i] - mean; var += d * d; }
  var = block_sum(var, red) * (1.f / 1024.f);
  rstd = rsqrtf(var + 1e-5f);
#pragma unroll
  for (int i = 0; i < 4; i++) {
    int idx = tid + 256 * i;
    out[(size_t)row * 1024 + idx] = (y[i] - mean) * rstd * g3[idx] + b3[idx];
  }
}

// ------------------------------ launcher -----------------------------------
extern "C" void kernel_launch(void* const* d_in, const int* in_sizes, int n_in,
                              void* d_out, int out_size, void* d_ws,
                              size_t ws_size, hipStream_t stream) {
  const float* x     = (const float*)d_in[0];
  const float* mask  = (const float*)d_in[1];
  const float* wq    = (const float*)d_in[2];
  const float* wk    = (const float*)d_in[3];
  const float* wv    = (const float*)d_in[4];
  const float* w1    = (const float*)d_in[5];
  const float* b1    = (const float*)d_in[6];
  const float* w2    = (const float*)d_in[7];
  const float* b2    = (const float*)d_in[8];
  const float* ln1s  = (const float*)d_in[9];
  const float* ln1b  = (const float*)d_in[10];
  const float* ln2s  = (const float*)d_in[11];
  const float* ln2b  = (const float*)d_in[12];
  const float* ln3s  = (const float*)d_in[13];
  const float* ln3b  = (const float*)d_in[14];
  float* out = (float*)d_out;

  // workspace carve-up (~111 MB, 256B aligned)
  char* ws = (char*)d_ws;
  size_t off = 0;
  auto alloc = [&](size_t bytes) {
    char* p = ws + off;
    off = (off + bytes + 255) & ~(size_t)255;
    return p;
  };
  const size_t ND2 = (size_t)NROW * D_ * 2;   // f16 activation
  const size_t ND4 = (size_t)NROW * D_ * 4;   // f32 activation
  const size_t WW2 = (size_t)D_ * D_ * 2;     // f16 weight
  _Float16* xh   = (_Float16*)alloc(ND2);
  _Float16* wqh  = (_Float16*)alloc(WW2);
  _Float16* wkh  = (_Float16*)alloc(WW2);
  _Float16* wvh  = (_Float16*)alloc(WW2);
  _Float16* w1h  = (_Float16*)alloc(WW2);
  _Float16* w2h  = (_Float16*)alloc(WW2);
  _Float16* Qh   = (_Float16*)alloc(ND2);
  _Float16* Kh   = (_Float16*)alloc(ND2);
  _Float16* Vh   = (_Float16*)alloc(ND2);
  _Float16* h1h  = (_Float16*)alloc(ND2);
  _Float16* f1h  = (_Float16*)alloc(ND2);
  float* attnout = (float*)alloc(ND4);
  float* h1f     = (float*)alloc(ND4);
  float* f2f     = (float*)alloc(ND4);

  const int NACT = NROW * D_ / 256;   // 16384 blocks
  const int NWGT = D_ * D_ / 256;     // 4096 blocks
  cvt_f32_f16<<<NACT, 256, 0, stream>>>(x,  xh);
  cvt_f32_f16<<<NWGT, 256, 0, stream>>>(wq, wqh);
  cvt_f32_f16<<<NWGT, 256, 0, stream>>>(wk, wkh);
  cvt_f32_f16<<<NWGT, 256, 0, stream>>>(wv, wvh);
  cvt_f32_f16<<<NWGT, 256, 0, stream>>>(w1, w1h);
  cvt_f32_f16<<<NWGT, 256, 0, stream>>>(w2, w2h);

  dim3 gg(D_ / 128, NROW / 64);  // (8, 64)
  // Q, K, V projections: no bias, f16 output only
  gemm_kernel<false, false, false, true><<<gg, 256, 0, stream>>>(
      xh, wqh, nullptr, nullptr, Qh);
  gemm_kernel<false, false, false, true><<<gg, 256, 0, stream>>>(
      xh, wkh, nullptr, nullptr, Kh);
  gemm_kernel<false, false, false, true><<<gg, 256, 0, stream>>>(
      xh, wvh, nullptr, nullptr, Vh);

  attn_kernel<<<dim3(T_ / 32, B_ * H_), 256, 0, stream>>>(Qh, Kh, Vh, attnout);

  ln1_kernel<<<NROW, 256, 0, stream>>>(x, attnout, mask, ln1s, ln1b, h1f, h1h);

  // FFN1: bias + relu, f16 out;  FFN2: bias, f32 out
  gemm_kernel<true, true, false, true><<<gg, 256, 0, stream>>>(
      h1h, w1h, b1, nullptr, f1h);
  gemm_kernel<true, false, true, false><<<gg, 256, 0, stream>>>(
      f1h, w2h, b2, f2f, nullptr);

  ln23_kernel<<<NROW, 256, 0, stream>>>(h1f, f2f, ln2s, ln2b, ln3s, ln3b, out);
  (void)in_sizes; (void)n_in; (void)out_size; (void)ws_size;
}
